// PNA_GNN_QM9_9328668966929
// MI455X (gfx1250) — matrix operand order repaired and checked by reading, not compiled, not run on hardware
//
#include <hip/hip_runtime.h>
#include <hip/hip_bf16.h>

// Problem constants (from reference): N=4096, E=65536, H=128, ED=64, L=3, T=4, FO=32
#define NN   4096
#define EE   65536
#define HH   128
#define EDD  64
#define LL   3
#define TT   4
#define FOO  32

typedef __attribute__((ext_vector_type(16))) __bf16 v16bf;
typedef __attribute__((ext_vector_type(8)))  float  v8f;
typedef unsigned long long u64;

// ---------- helpers ----------
__device__ __forceinline__ __bf16 f2bf(float f) {
  unsigned u = __builtin_bit_cast(unsigned, f);
  unsigned short h = (unsigned short)((u + 0x7FFFu + ((u >> 16) & 1u)) >> 16); // RNE
  return __builtin_bit_cast(__bf16, h);
}
__device__ __forceinline__ unsigned okey(float v) {         // order-preserving key for atomicMax
  unsigned u = __builtin_bit_cast(unsigned, v);
  return (u & 0x80000000u) ? ~u : (u | 0x80000000u);
}
__device__ __forceinline__ float dekey(unsigned k) {
  unsigned u = (k & 0x80000000u) ? (k & 0x7FFFFFFFu) : ~k;
  return __builtin_bit_cast(float, u);
}

__device__ __forceinline__ v16bf ld_frag_lds(const __bf16* p) {
  union { uint4 u[2]; v16bf v; } r;
  r.u[0] = *(const uint4*)(p);
  r.u[1] = *(const uint4*)(p + 16);
  return r.v;
}
__device__ __forceinline__ v16bf ld_frag_w(const __bf16* p) {
  union { uint4 u[2]; v16bf v; } r;
  r.u[0] = *(const uint4*)(p);
  r.u[1] = *(const uint4*)(p + 16);
  return r.v;
}

// ---------- generic bf16 WMMA GEMM:  C[M,Nc] = act( A[M,K] * W[Nc,K]^T + bias (+ beta*C) )
// A rows optionally gathered via rowIdx. Writes f32 C (Cf) and/or bf16 C (Cb).
// Block: 128 threads = 4 waves; block tile = 16 M rows x 256 N cols.
// Each wave owns a 16x64 slab (4 accumulators) -> 4 WMMAs per 32-wide K step
// off one shared A fragment. A tile (16 x K) staged to LDS once, one barrier.
// Validity of the wave's 4 N-tiles is decided ONCE before the K loop so the
// dominant path has no exec-mask traffic around WMMA (EXEC all-ones).
template<int K>
__global__ __launch_bounds__(128)
void gemm_bf16_k(const __bf16* __restrict__ A, int lda, const int* __restrict__ rowIdx,
                 const __bf16* __restrict__ W, int ldw,
                 const float* __restrict__ bias,
                 float* __restrict__ Cf, int ldc,
                 __bf16* __restrict__ Cb, int ldcb,
                 int Nc, int beta, int act)
{
  __shared__ __align__(16) __bf16 As[16 * K];
  const int tid  = threadIdx.x;
  const int wave = tid >> 5;
  const int lane = tid & 31;
  const int m0 = blockIdx.x * 16;

  // ---- stage A tile to LDS (whole K), one barrier ----
  {
    const int sRow = tid >> 3;                    // 0..15
    const int sCol = (tid & 7) * 4;               // 0,4,...,28
    int srcRow = m0 + sRow;
    if (rowIdx) srcRow = rowIdx[srcRow];
    const __bf16* aRow = A + (long)srcRow * lda;
    __bf16* dRow = &As[sRow * K];
#pragma unroll
    for (int k0 = 0; k0 < K; k0 += 32)
      *(u64*)(dRow + k0 + sCol) = *(const u64*)(aRow + k0 + sCol);
  }
  __syncthreads();

  // ---- compute: wave covers n0w .. n0w+63 (4 tiles of 16) ----
  const int n0w = blockIdx.y * 256 + wave * 64;
  if (n0w >= Nc) return;
  const int nl  = lane & 15;
  const int kaA = (lane < 16) ? 0 : 8;            // A frag: lanes 0-15 K{0..7,16..23}; 16-31 K{8..15,24..31}
  const int kaB = (lane < 16) ? 0 : 8;            // B frag: lane = out col, K{0..7,16..23} / K{8..15,24..31}
  const __bf16* aF = &As[nl * K + kaA];

  v8f acc0 = {}, acc1 = {}, acc2 = {}, acc3 = {};
  const __bf16* w0 = W + (long)(n0w + nl) * ldw + kaB;
  const long wstep = (long)16 * ldw;
  int nvalid;

  if (n0w + 64 <= Nc) {
    // -------- fast path: all 4 tiles valid, no branches in the loop --------
    nvalid = 4;
#pragma unroll 4
    for (int k0 = 0; k0 < K; k0 += 32) {
      const v16bf a = ld_frag_lds(aF + k0);
      acc0 = __builtin_amdgcn_wmma_f32_16x16x32_bf16(false, a, false,
               ld_frag_w(w0 + k0), (short)0, acc0, false, false);
      acc1 = __builtin_amdgcn_wmma_f32_16x16x32_bf16(false, a, false,
               ld_frag_w(w0 + wstep + k0), (short)0, acc1, false, false);
      acc2 = __builtin_amdgcn_wmma_f32_16x16x32_bf16(false, a, false,
               ld_frag_w(w0 + 2 * wstep + k0), (short)0, acc2, false, false);
      acc3 = __builtin_amdgcn_wmma_f32_16x16x32_bf16(false, a, false,
               ld_frag_w(w0 + 3 * wstep + k0), (short)0, acc3, false, false);
    }
  } else {
    // -------- slow path: trailing partial wave (only Nc%64 != 0 cases) --------
    nvalid = (Nc - n0w + 15) >> 4;                // 1..3 valid tiles
    for (int k0 = 0; k0 < K; k0 += 32) {
      const v16bf a = ld_frag_lds(aF + k0);
      acc0 = __builtin_amdgcn_wmma_f32_16x16x32_bf16(false, a, false,
               ld_frag_w(w0 + k0), (short)0, acc0, false, false);
      if (nvalid > 1)
        acc1 = __builtin_amdgcn_wmma_f32_16x16x32_bf16(false, a, false,
                 ld_frag_w(w0 + wstep + k0), (short)0, acc1, false, false);
      if (nvalid > 2)
        acc2 = __builtin_amdgcn_wmma_f32_16x16x32_bf16(false, a, false,
                 ld_frag_w(w0 + 2 * wstep + k0), (short)0, acc2, false, false);
    }
  }

  // ---- epilogue ----
  v8f* accs[4] = { &acc0, &acc1, &acc2, &acc3 };
#pragma unroll
  for (int j = 0; j < 4; ++j) {
    if (j >= nvalid) break;
    const int ncol = n0w + j * 16 + nl;
    const float bv = bias ? bias[ncol] : 0.0f;
    v8f a = *accs[j];
#pragma unroll
    for (int i = 0; i < 8; ++i) {
      const int r = m0 + ((lane < 16) ? i : 8 + i); // C/D layout: lane=N, VGPR i = M row
      float v = a[i] + bv;
      if (beta && Cf) v += Cf[(long)r * ldc + ncol];
      if (act) v = v > 0.0f ? v : 0.0f;
      if (Cf) Cf[(long)r * ldc + ncol] = v;
      if (Cb) Cb[(long)r * ldcb + ncol] = f2bf(v);
    }
  }
}

// ---------- weight folding: W3p[l,r,k] = sum_f preW[l,r,256+f] * encW[l,f,k] ----------
__global__ void fold_w3(const float* __restrict__ preW, const float* __restrict__ encW,
                        float* __restrict__ W3p)
{
  int idx = blockIdx.x * blockDim.x + threadIdx.x;
  if (idx >= LL * 512 * 64) return;
  int k = idx & 63, r = (idx >> 6) & 511, l = idx >> 15;
  const float* pw = preW + ((long)l * 512 + r) * 384 + 256;
  const float* ew = encW + (long)l * 128 * 64 + k;
  float s = 0.0f;
  for (int f = 0; f < 128; ++f) s += pw[f] * ew[(long)f * 64];
  W3p[idx] = s;
}
__global__ void fold_b(const float* __restrict__ preW, const float* __restrict__ preB,
                       const float* __restrict__ encB, float* __restrict__ bP)
{
  int idx = blockIdx.x * blockDim.x + threadIdx.x;
  if (idx >= LL * 512) return;
  int r = idx & 511, l = idx >> 9;
  const float* pw = preW + ((long)l * 512 + r) * 384 + 256;
  const float* eb = encB + l * 128;
  float s = preB[idx];
  for (int f = 0; f < 128; ++f) s += pw[f] * eb[f];
  bP[idx] = s;
}

// ---------- misc elementwise kernels ----------
__global__ void cast_slice(const float* __restrict__ src, __bf16* __restrict__ dst,
                           int R, int C, int ldsrc, int scol, int lddst)
{
  int idx = blockIdx.x * blockDim.x + threadIdx.x;
  if (idx >= R * C) return;
  int r = idx / C, c = idx - r * C;
  dst[(long)r * lddst + c] = f2bf(src[(long)r * ldsrc + scol + c]);
}
__global__ void init_cast(const float* __restrict__ src, float* __restrict__ dF,
                          __bf16* __restrict__ dB, long n)
{
  long idx = blockIdx.x * (long)blockDim.x + threadIdx.x;
  if (idx >= n) return;
  float v = src[idx];
  dF[idx] = v; dB[idx] = f2bf(v);
}
__global__ void zero_agg(float* __restrict__ s, float* __restrict__ s2,
                         unsigned* __restrict__ mx, long n)
{
  long idx = blockIdx.x * (long)blockDim.x + threadIdx.x;
  if (idx >= n) return;
  s[idx] = 0.0f; s2[idx] = 0.0f; mx[idx] = 0u;
}
__global__ void zero_i32(int* __restrict__ p, int n)
{
  int idx = blockIdx.x * blockDim.x + threadIdx.x;
  if (idx < n) p[idx] = 0;
}
__global__ void count_edges(const int* __restrict__ dst, int* __restrict__ cnt, int e)
{
  int idx = blockIdx.x * blockDim.x + threadIdx.x;
  if (idx < e) atomicAdd(&cnt[dst[idx]], 1);
}
__global__ void copy_f32(const float* __restrict__ src, float* __restrict__ dst, long n)
{
  long idx = blockIdx.x * (long)blockDim.x + threadIdx.x;
  if (idx < n) dst[idx] = src[idx];
}

// ---------- segment aggregation over edges (sum / sum^2 / max keyed by dst) ----------
__global__ void aggregate(const float* __restrict__ msg, const int* __restrict__ dst,
                          float* __restrict__ sum, float* __restrict__ sumsq,
                          unsigned* __restrict__ mxI)
{
  long idx = blockIdx.x * (long)blockDim.x + threadIdx.x;
  if (idx >= (long)EE * 512) return;
  int e = (int)(idx >> 9), c = (int)(idx & 511);
  float v = msg[idx];
  long o = (long)dst[e] * 512 + c;
  atomicAdd(&sum[o], v);
  atomicAdd(&sumsq[o], v * v);
  atomicMax(&mxI[o], okey(v));
}

// ---------- finalize PNA aggregation -> cat[N, T, 512] = [x | max | std | mean] (bf16) ----
__global__ void finalize_agg(const float* __restrict__ sum, const float* __restrict__ sumsq,
                             const unsigned* __restrict__ mxI, const int* __restrict__ cnt,
                             const float* __restrict__ node, __bf16* __restrict__ cat)
{
  long idx = blockIdx.x * (long)blockDim.x + threadIdx.x;
  if (idx >= (long)NN * 512) return;
  int n = (int)(idx >> 9), tf = (int)(idx & 511);
  int t = tf >> 7, f = tf & 127;
  float c = (float)cnt[n];
  float denom = c > 1.0f ? c : 1.0f;
  float mean = sum[idx] / denom;
  float m2   = sumsq[idx] / denom;
  float var  = m2 - mean * mean; var = var > 0.0f ? var : 0.0f;
  float sd   = sqrtf(var + 1e-5f);
  float mx   = (c > 0.0f) ? dekey(mxI[idx]) : 0.0f;
  __bf16* base = cat + (long)n * 2048 + (long)t * 512;
  base[f]       = f2bf(node[(long)n * 128 + f]);
  base[128 + f] = f2bf(mx);
  base[256 + f] = f2bf(sd);
  base[384 + f] = f2bf(mean);
}

// ---------- GRU gates (gi, gh precomputed via WMMA GEMMs) ----------
__global__ void gru_gate(const float* __restrict__ gi, const float* __restrict__ gh,
                         float* __restrict__ nodeF, __bf16* __restrict__ nodeB)
{
  long idx = blockIdx.x * (long)blockDim.x + threadIdx.x;
  if (idx >= (long)NN * 128) return;
  int n = (int)(idx >> 7), c = (int)(idx & 127);
  const float* gin = gi + (long)n * 384;
  const float* ghn = gh + (long)n * 384;
  float r = 1.0f / (1.0f + expf(-(gin[c] + ghn[c])));
  float z = 1.0f / (1.0f + expf(-(gin[128 + c] + ghn[128 + c])));
  float cc = tanhf(gin[256 + c] + r * ghn[256 + c]);
  float h = nodeF[idx];
  float hn = (1.0f - z) * cc + z * h;
  nodeF[idx] = hn; nodeB[idx] = f2bf(hn);
}

// ---------- edge-update gathers ----------
__global__ void gather_uv(const __bf16* __restrict__ nodeB, const int* __restrict__ src,
                          const int* __restrict__ dst, __bf16* __restrict__ uv)
{
  long idx = blockIdx.x * (long)blockDim.x + threadIdx.x;
  if (idx >= (long)EE * 256) return;
  int e = (int)(idx >> 8), c = (int)(idx & 255);
  int nd = (c < 128) ? src[e] : dst[e];            // uv = [out[src] | out[dst]]
  uv[idx] = nodeB[(long)nd * 128 + (c & 127)];
}
__global__ void copy_ea_cat2(const __bf16* __restrict__ eaB, __bf16* __restrict__ cat2)
{
  long idx = blockIdx.x * (long)blockDim.x + threadIdx.x;
  if (idx >= (long)EE * 64) return;
  int e = (int)(idx >> 6), c = (int)(idx & 63);
  cat2[(long)e * 192 + 128 + c] = eaB[idx];
}

// ---------- host-side launch helpers ----------
static inline void gemm(hipStream_t s, const __bf16* A, int lda, const int* rowIdx,
                        const __bf16* W, int ldw, const float* bias,
                        float* Cf, int ldc, __bf16* Cb, int ldcb,
                        int M, int Nc, int K, int beta, int act)
{
  dim3 g(M / 16, (Nc + 255) / 256);
  dim3 b(128);
  switch (K) {
    case 64:
      gemm_bf16_k<64><<<g, b, 0, s>>>(A, lda, rowIdx, W, ldw, bias, Cf, ldc, Cb, ldcb, Nc, beta, act); break;
    case 128:
      gemm_bf16_k<128><<<g, b, 0, s>>>(A, lda, rowIdx, W, ldw, bias, Cf, ldc, Cb, ldcb, Nc, beta, act); break;
    case 192:
      gemm_bf16_k<192><<<g, b, 0, s>>>(A, lda, rowIdx, W, ldw, bias, Cf, ldc, Cb, ldcb, Nc, beta, act); break;
    case 256:
      gemm_bf16_k<256><<<g, b, 0, s>>>(A, lda, rowIdx, W, ldw, bias, Cf, ldc, Cb, ldcb, Nc, beta, act); break;
    default:
      gemm_bf16_k<512><<<g, b, 0, s>>>(A, lda, rowIdx, W, ldw, bias, Cf, ldc, Cb, ldcb, Nc, beta, act); break;
  }
}
static inline int blks(long n, int b) { return (int)((n + b - 1) / b); }

extern "C" void kernel_launch(void* const* d_in, const int* in_sizes, int n_in,
                              void* d_out, int out_size, void* d_ws, size_t ws_size,
                              hipStream_t stream)
{
  (void)in_sizes; (void)n_in; (void)out_size; (void)ws_size;
  const float* x       = (const float*)d_in[0];
  const float* ea_in   = (const float*)d_in[1];
  const int*   eidx    = (const int*)d_in[2];
  const float* enc_W   = (const float*)d_in[3];
  const float* enc_b   = (const float*)d_in[4];
  const float* pre_W   = (const float*)d_in[5];
  const float* pre_b   = (const float*)d_in[6];
  const float* post_W  = (const float*)d_in[7];
  const float* post_b  = (const float*)d_in[8];
  const float* lin_W   = (const float*)d_in[9];
  const float* lin_b   = (const float*)d_in[10];
  const float* gWih    = (const float*)d_in[11];
  const float* gWhh    = (const float*)d_in[12];
  const float* gbih    = (const float*)d_in[13];
  const float* gbhh    = (const float*)d_in[14];
  const float* em1W    = (const float*)d_in[15];
  const float* em1b    = (const float*)d_in[16];
  const float* em2W    = (const float*)d_in[17];
  const float* em2b    = (const float*)d_in[18];
  const float* eu1W    = (const float*)d_in[19];
  const float* eu1b    = (const float*)d_in[20];
  const float* eu2W    = (const float*)d_in[21];
  const float* eu2b    = (const float*)d_in[22];
  const int* srcI = eidx;
  const int* dstI = eidx + EE;
  float* out = (float*)d_out;

  // ---- workspace carve-out ----
  char* base = (char*)d_ws; size_t off = 0;
  auto alloc = [&](size_t bytes) -> void* {
    off = (off + 255) & ~(size_t)255;
    void* p = base + off; off += bytes; return p;
  };
  float*    nodeF = (float*)   alloc((size_t)NN * 128 * 4);
  __bf16*   nodeB = (__bf16*)  alloc((size_t)NN * 128 * 2);
  float*    eaF   = (float*)   alloc((size_t)EE * 64 * 4);
  __bf16*   eaB   = (__bf16*)  alloc((size_t)EE * 64 * 2);
  int*      cnt   = (int*)     alloc((size_t)NN * 4);
  float*    aggS  = (float*)   alloc((size_t)NN * 512 * 4);
  float*    aggQ  = (float*)   alloc((size_t)NN * 512 * 4);
  unsigned* aggM  = (unsigned*)alloc((size_t)NN * 512 * 4);
  __bf16*   catB  = (__bf16*)  alloc((size_t)NN * 2048 * 2);
  __bf16*   yB    = (__bf16*)  alloc((size_t)NN * 128 * 2);
  __bf16*   mB    = (__bf16*)  alloc((size_t)NN * 128 * 2);
  float*    giF   = (float*)   alloc((size_t)NN * 384 * 4);
  float*    ghF   = (float*)   alloc((size_t)NN * 384 * 4);
  // folded + bf16 weights
  float*    W3p   = (float*)   alloc((size_t)LL * 512 * 64 * 4);
  float*    bP    = (float*)   alloc((size_t)LL * 512 * 4);
  __bf16*   W3pB  = (__bf16*)  alloc((size_t)LL * 512 * 64 * 2);
  __bf16*   W1B   = (__bf16*)  alloc((size_t)LL * 512 * 128 * 2);
  __bf16*   W2B   = (__bf16*)  alloc((size_t)LL * 512 * 128 * 2);
  __bf16*   qWB   = (__bf16*)  alloc((size_t)LL * 128 * 512 * 2);
  __bf16*   lWB   = (__bf16*)  alloc((size_t)LL * 128 * 128 * 2);
  __bf16*   WihB  = (__bf16*)  alloc((size_t)384 * 128 * 2);
  __bf16*   WhhB  = (__bf16*)  alloc((size_t)384 * 128 * 2);
  __bf16*   em1B  = (__bf16*)  alloc((size_t)256 * 256 * 2);
  __bf16*   em2B  = (__bf16*)  alloc((size_t)128 * 256 * 2);
  __bf16*   eu1B  = (__bf16*)  alloc((size_t)128 * 192 * 2);
  __bf16*   eu2B  = (__bf16*)  alloc((size_t)64 * 128 * 2);
  // big aliased region: msg[E,512] f32, later reused for edge-MLP intermediates
  char*   big  = (char*)alloc((size_t)EE * 512 * 4);
  float*  msg  = (float*)big;
  __bf16* uvB  = (__bf16*)big;                                  // E*256*2 = 32MiB
  __bf16* m1B  = (__bf16*)(big + (size_t)EE * 256 * 2);         // E*256*2
  __bf16* cat2 = (__bf16*)(big + (size_t)EE * 256 * 4);         // E*192*2
  __bf16* tB   = (__bf16*)(big + (size_t)EE * 256 * 4 + (size_t)EE * 192 * 2); // E*128*2

  const int TB = 256;

  // ---- one-time prep ----
  fold_w3<<<blks(LL * 512 * 64, TB), TB, 0, stream>>>(pre_W, enc_W, W3p);
  fold_b <<<blks(LL * 512, TB), TB, 0, stream>>>(pre_W, pre_b, enc_b, bP);
  cast_slice<<<blks(LL * 512 * 64, TB), TB, 0, stream>>>(W3p, W3pB, LL * 512, 64, 64, 0, 64);
  for (int l = 0; l < LL; ++l) {
    cast_slice<<<blks(512 * 128, TB), TB, 0, stream>>>(pre_W + (size_t)l * 512 * 384, W1B + (size_t)l * 512 * 128, 512, 128, 384, 0,   128);
    cast_slice<<<blks(512 * 128, TB), TB, 0, stream>>>(pre_W + (size_t)l * 512 * 384, W2B + (size_t)l * 512 * 128, 512, 128, 384, 128, 128);
    cast_slice<<<blks(128 * 512, TB), TB, 0, stream>>>(post_W + (size_t)l * 128 * 512, qWB + (size_t)l * 128 * 512, 128, 512, 512, 0, 512);
    cast_slice<<<blks(128 * 128, TB), TB, 0, stream>>>(lin_W + (size_t)l * 128 * 128, lWB + (size_t)l * 128 * 128, 128, 128, 128, 0, 128);
  }
  cast_slice<<<blks(384 * 128, TB), TB, 0, stream>>>(gWih, WihB, 384, 128, 128, 0, 128);
  cast_slice<<<blks(384 * 128, TB), TB, 0, stream>>>(gWhh, WhhB, 384, 128, 128, 0, 128);
  cast_slice<<<blks(256 * 256, TB), TB, 0, stream>>>(em1W, em1B, 256, 256, 256, 0, 256);
  cast_slice<<<blks(128 * 256, TB), TB, 0, stream>>>(em2W, em2B, 128, 256, 256, 0, 256);
  cast_slice<<<blks(128 * 192, TB), TB, 0, stream>>>(eu1W, eu1B, 128, 192, 192, 0, 192);
  cast_slice<<<blks(64 * 128, TB), TB, 0, stream>>>(eu2W, eu2B, 64, 128, 128, 0, 128);

  init_cast<<<blks((long)NN * 128, TB), TB, 0, stream>>>(x, nodeF, nodeB, (long)NN * 128);
  init_cast<<<blks((long)EE * 64, TB), TB, 0, stream>>>(ea_in, eaF, eaB, (long)EE * 64);
  zero_i32<<<blks(NN, TB), TB, 0, stream>>>(cnt, NN);
  count_edges<<<blks(EE, TB), TB, 0, stream>>>(dstI, cnt, EE);

  // ---- layers ----
  for (int l = 0; l < LL; ++l) {
    // PNA message: msg = x[dst]*W1^T + x[src]*W2^T + ea*W3p^T + b'
    gemm(stream, nodeB, 128, dstI, W1B + (size_t)l * 512 * 128, 128, nullptr,
         msg, 512, nullptr, 0, EE, 512, 128, 0, 0);
    gemm(stream, nodeB, 128, srcI, W2B + (size_t)l * 512 * 128, 128, nullptr,
         msg, 512, nullptr, 0, EE, 512, 128, 1, 0);
    gemm(stream, eaB, 64, nullptr, W3pB + (size_t)l * 512 * 64, 64, bP + (size_t)l * 512,
         msg, 512, nullptr, 0, EE, 512, 64, 1, 0);

    // segment aggregation
    zero_agg<<<blks((long)NN * 512, TB), TB, 0, stream>>>(aggS, aggQ, aggM, (long)NN * 512);
    aggregate<<<blks((long)EE * 512, TB), TB, 0, stream>>>(msg, dstI, aggS, aggQ, aggM);
    finalize_agg<<<blks((long)NN * 512, TB), TB, 0, stream>>>(aggS, aggQ, aggM, cnt, nodeF, catB);

    // per-tower post-MLP: y[:, t*32 : t*32+32]
    for (int t = 0; t < TT; ++t)
      gemm(stream, catB + (size_t)t * 512, 2048, nullptr,
           qWB + (size_t)l * 128 * 512 + (size_t)t * 32 * 512, 512,
           post_b + ((size_t)l * TT + t) * 32,
           nullptr, 0, yB + (size_t)t * 32, 128, NN, 32, 512, 0, 0);

    // m = relu(y * lW^T + lb)
    gemm(stream, yB, 128, nullptr, lWB + (size_t)l * 128 * 128, 128, lin_b + (size_t)l * 128,
         nullptr, 0, mB, 128, NN, 128, 128, 0, 1);

    // GRU: gi = m*Wih^T+bih ; gh = h*Whh^T+bhh ; gates
    gemm(stream, mB,    128, nullptr, WihB, 128, gbih, giF, 384, nullptr, 0, NN, 384, 128, 0, 0);
    gemm(stream, nodeB, 128, nullptr, WhhB, 128, gbhh, ghF, 384, nullptr, 0, NN, 384, 128, 0, 0);
    gru_gate<<<blks((long)NN * 128, TB), TB, 0, stream>>>(giF, ghF, nodeF, nodeB);

    // edge update MLP chain (msg region now reusable)
    gather_uv<<<blks((long)EE * 256, TB), TB, 0, stream>>>(nodeB, srcI, dstI, uvB);
    gemm(stream, uvB, 256, nullptr, em1B, 256, em1b, nullptr, 0, m1B, 256, EE, 256, 256, 0, 1);
    gemm(stream, m1B, 256, nullptr, em2B, 256, em2b, nullptr, 0, cat2, 192, EE, 128, 256, 0, 1);
    copy_ea_cat2<<<blks((long)EE * 64, TB), TB, 0, stream>>>(eaB, cat2);
    gemm(stream, cat2, 192, nullptr, eu1B, 192, eu1b, nullptr, 0, tB, 128, EE, 128, 192, 0, 1);
    gemm(stream, tB, 128, nullptr, eu2B, 128, eu2b, eaF, 64, eaB, 64, EE, 64, 128, 0, 0);
  }

  // ---- outputs: (out, ea, out) ----
  copy_f32<<<blks((long)NN * 128, TB), TB, 0, stream>>>(nodeF, out, (long)NN * 128);
  copy_f32<<<blks((long)EE * 64, TB), TB, 0, stream>>>(eaF, out + (size_t)NN * 128, (long)EE * 64);
  copy_f32<<<blks((long)NN * 128, TB), TB, 0, stream>>>(nodeF, out + (size_t)NN * 128 + (size_t)EE * 64, (long)NN * 128);
}
